// VectorQuantize_61959198212260
// MI455X (gfx1250) — matrix operand order repaired and checked by reading, not compiled, not run on hardware
//
#include <hip/hip_runtime.h>
#include <hip/hip_bf16.h>

// ---------------------------------------------------------------------------
// VQ-VAE vector-quantize forward, fused for MI455X (gfx1250, wave32, WMMA).
// B=8 N=1024 DIM=256 CDIM=64 K=16384.  Dominant cost: 8192x16384x64 distance
// GEMM fused with argmin (bf16 WMMA, f32 accumulate).  ||zp||^2 cancels in
// the argmin, so only enorm[k] - 2*dot is scored.
// Round 2: 32-row blocks (each B fragment feeds 2 WMMAs), branch-free
// shfl_xor argmin reduction.
// ---------------------------------------------------------------------------

typedef __attribute__((ext_vector_type(16))) __bf16 v16bf;
typedef __attribute__((ext_vector_type(8)))  __bf16 v8bf;
typedef __attribute__((ext_vector_type(8)))  float  v8f;

#define NROWS   8192
#define CD      64
#define KCODES  16384
#define DIMSZ   256

static __device__ __forceinline__ unsigned short f2bf(float f) {
    unsigned int u = __float_as_uint(f);
    unsigned int r = u + 0x7FFFu + ((u >> 16) & 1u);   // round-to-nearest-even
    return (unsigned short)(r >> 16);
}

// branch-free (dist, idx) argmin select with first-index tie-break
static __device__ __forceinline__ void selmin(float& bd, int& bi, float d, int i) {
    bool take = (d < bd) | ((d == bd) & (i < bi));
    bd = take ? d : bd;
    bi = take ? i : bi;
}

// --------------------------------------------------------------------------
// K0: emb -> bf16 copy + per-code squared norm; zero the loss accumulator.
// --------------------------------------------------------------------------
__global__ __launch_bounds__(256) void vq_prep_emb(const float* __restrict__ emb,
                                                   unsigned short* __restrict__ embB,
                                                   float* __restrict__ enorm,
                                                   float* __restrict__ lossAcc) {
    int k = blockIdx.x * blockDim.x + threadIdx.x;
    if (k == 0) *lossAcc = 0.0f;
    if (k < KCODES) {
        float s = 0.0f;
        #pragma unroll 8
        for (int c = 0; c < CD; ++c) {
            float v = emb[k * CD + c];
            s = fmaf(v, v, s);
            embB[k * CD + c] = f2bf(v);
        }
        enorm[k] = s;
    }
}

// --------------------------------------------------------------------------
// K1: zp = z @ Wq^T + bq  (fp32, plus bf16 copy for the WMMA distance GEMM).
// Block = 256 threads = 4 rows x 64 output channels; z rows staged in LDS.
// --------------------------------------------------------------------------
__global__ __launch_bounds__(256) void vq_proj_q(const float* __restrict__ z,
                                                 const float* __restrict__ Wq,
                                                 const float* __restrict__ bq,
                                                 float* __restrict__ zp,
                                                 unsigned short* __restrict__ zpB) {
    __shared__ float zsh[4][DIMSZ];
    const int rb = blockIdx.x * 4;
    for (int i = threadIdx.x; i < 4 * DIMSZ; i += 256)
        zsh[i >> 8][i & 255] = z[(size_t)(rb + (i >> 8)) * DIMSZ + (i & 255)];
    __syncthreads();

    const int r = threadIdx.x >> 6;   // local row 0..3
    const int c = threadIdx.x & 63;   // output channel 0..63
    const float* wq = Wq + (size_t)c * DIMSZ;
    float s = 0.0f;
    #pragma unroll 8
    for (int k = 0; k < DIMSZ; ++k) s = fmaf(zsh[r][k], wq[k], s);
    s += bq[c];
    const int row = rb + r;
    zp [row * CD + c] = s;
    zpB[row * CD + c] = f2bf(s);
}

// --------------------------------------------------------------------------
// K2: fused distance + argmin.
// Block = 32 rows (two 16-row groups), 8 waves.  Each wave owns code-tiles
// t = wave + 8*i (16 codes).  Per tile: one B-fragment pair feeds FOUR
// v_wmma_f32_16x16x32_bf16 (2 row groups x K=64), then
// score = enorm[n] - 2*dot, running per-lane argmin, shfl_xor column reduce,
// LDS cross-wave reduce.
//
// A fragment (16x32 bf16): lane L<16 -> row L, K 0..15 contiguous;
//   lane L+16 -> row L, K 16..31 contiguous.
// B fragment (32x16 bf16): lane L<16 -> col L, K 0..15 contiguous;
//   lane L+16 -> col L-16, K 16..31 contiguous.
// C/D (16x16 f32): VGPR r: lanes0-15 = (M=r, N=lane); lanes16-31 = (M=r+8).
// --------------------------------------------------------------------------
__global__ __launch_bounds__(256) void vq_argmin(const unsigned short* __restrict__ zpB,
                                                 const unsigned short* __restrict__ embB,
                                                 const float* __restrict__ enorm,
                                                 int* __restrict__ idxOut,
                                                 float* __restrict__ idxOutF) {
    const int wave = threadIdx.x >> 5;
    const int lane = threadIdx.x & 31;
    const int half = lane >> 4;      // 0 = lanes 0-15, 1 = lanes 16-31
    const int l16  = lane & 15;
    const int m0   = blockIdx.x * 32;

    // ---- A fragments: zp rows m0..m0+31 as two 16-row groups, K=0..63 ----
    v16bf a0k0, a0k1, a1k0, a1k1;
    {
        const v8bf* zr0 = reinterpret_cast<const v8bf*>(zpB + (size_t)(m0 + l16) * CD);
        const v8bf* zr1 = reinterpret_cast<const v8bf*>(zpB + (size_t)(m0 + 16 + l16) * CD);
        v8bf q0 = zr0[2 * half + 0], q1 = zr0[2 * half + 1], q2 = zr0[2 * half + 4], q3 = zr0[2 * half + 5];
        v8bf r0 = zr1[2 * half + 0], r1 = zr1[2 * half + 1], r2 = zr1[2 * half + 4], r3 = zr1[2 * half + 5];
        #pragma unroll
        for (int i = 0; i < 8; ++i) {
            a0k0[i] = q0[i]; a0k0[8 + i] = q1[i];
            a0k1[i] = q2[i]; a0k1[8 + i] = q3[i];
            a1k0[i] = r0[i]; a1k0[8 + i] = r1[i];
            a1k1[i] = r2[i]; a1k1[8 + i] = r3[i];
        }
    }

    float bd0[8], bd1[8];
    int   bi0[8], bi1[8];
    #pragma unroll
    for (int r = 0; r < 8; ++r) {
        bd0[r] = 3.4e38f; bi0[r] = 0x7fffffff;
        bd1[r] = 3.4e38f; bi1[r] = 0x7fffffff;
    }

    for (int t = wave; t < KCODES / 16; t += 8) {
        const int n = t * 16 + l16;                        // this lane's column/code

        const unsigned short* ecol = embB + (size_t)n * CD + half * 16;
        // prefetch this wave's next tile (8 tiles = 128 codes ahead)
        __builtin_prefetch(ecol + 128 * CD, 0, 0);

        v16bf b0 = *reinterpret_cast<const v16bf*>(ecol);        // K 0..15  / 16..31
        v16bf b1 = *reinterpret_cast<const v16bf*>(ecol + 32);   // K 32..47 / 48..63

        v8f acc0 = {0.f, 0.f, 0.f, 0.f, 0.f, 0.f, 0.f, 0.f};
        v8f acc1 = {0.f, 0.f, 0.f, 0.f, 0.f, 0.f, 0.f, 0.f};
        acc0 = __builtin_amdgcn_wmma_f32_16x16x32_bf16(false, a0k0, false, b0,
                                                       (short)0, acc0, false, false);
        acc1 = __builtin_amdgcn_wmma_f32_16x16x32_bf16(false, a1k0, false, b0,
                                                       (short)0, acc1, false, false);
        acc0 = __builtin_amdgcn_wmma_f32_16x16x32_bf16(false, a0k1, false, b1,
                                                       (short)0, acc0, false, false);
        acc1 = __builtin_amdgcn_wmma_f32_16x16x32_bf16(false, a1k1, false, b1,
                                                       (short)0, acc1, false, false);

        const float en = enorm[n];
        #pragma unroll
        for (int r = 0; r < 8; ++r) {
            float d0 = fmaf(-2.0f, acc0[r], en);     // ||zp||^2 cancels in argmin
            float d1 = fmaf(-2.0f, acc1[r], en);
            if (d0 < bd0[r]) { bd0[r] = d0; bi0[r] = n; }   // codes increase -> first-min
            if (d1 < bd1[r]) { bd1[r] = d1; bi1[r] = n; }
        }
    }

    // ---- column reduce: min over the 16 lanes of each half (masks 8,4,2,1) ----
    #pragma unroll
    for (int m = 8; m >= 1; m >>= 1) {
        #pragma unroll
        for (int r = 0; r < 8; ++r) {
            float od0 = __shfl_xor(bd0[r], m, 32);
            int   oi0 = __shfl_xor(bi0[r], m, 32);
            selmin(bd0[r], bi0[r], od0, oi0);
            float od1 = __shfl_xor(bd1[r], m, 32);
            int   oi1 = __shfl_xor(bi1[r], m, 32);
            selmin(bd1[r], bi1[r], od1, oi1);
        }
    }

    // ---- cross-wave reduce via LDS ----
    __shared__ float sD[8][32];
    __shared__ int   sI[8][32];
    if (l16 == 0) {
        #pragma unroll
        for (int r = 0; r < 8; ++r) {       // group g rows: g*16 + half*8 + r
            sD[wave][half * 8 + r]      = bd0[r];
            sI[wave][half * 8 + r]      = bi0[r];
            sD[wave][16 + half * 8 + r] = bd1[r];
            sI[wave][16 + half * 8 + r] = bi1[r];
        }
    }
    __syncthreads();
    if (threadIdx.x < 32) {
        float best  = sD[0][threadIdx.x];
        int   bestI = sI[0][threadIdx.x];
        #pragma unroll
        for (int w = 1; w < 8; ++w)
            selmin(best, bestI, sD[w][threadIdx.x], sI[w][threadIdx.x]);
        idxOut [m0 + threadIdx.x] = bestI;
        idxOutF[m0 + threadIdx.x] = (float)bestI;
    }
}

// --------------------------------------------------------------------------
// K3: out[row,:] = emb[idx[row]] @ Wp^T + bp   (fp32 gather + 64-FMA dot),
//     plus commit-loss partial sum (sum (z_q - zp)^2) via atomicAdd.
// --------------------------------------------------------------------------
__global__ __launch_bounds__(256) void vq_out(const float* __restrict__ emb,
                                              const float* __restrict__ Wp,
                                              const float* __restrict__ bp,
                                              const float* __restrict__ zp,
                                              const int* __restrict__ idx,
                                              float* __restrict__ out,
                                              float* __restrict__ lossAcc) {
    __shared__ float zq[CD];
    __shared__ float df[CD];
    const int row  = blockIdx.x;
    const int code = idx[row];
    if (threadIdx.x < CD) {
        float v = emb[(size_t)code * CD + threadIdx.x];
        zq[threadIdx.x] = v;
        float d = v - zp[(size_t)row * CD + threadIdx.x];
        df[threadIdx.x] = d * d;
    }
    __syncthreads();

    const int dch = threadIdx.x;                 // output channel 0..255
    const float* wp = Wp + (size_t)dch * CD;
    float s = 0.0f;
    #pragma unroll
    for (int c = 0; c < CD; ++c) s = fmaf(zq[c], wp[c], s);
    out[(size_t)row * DIMSZ + dch] = s + bp[dch];

    if (threadIdx.x == 0) {
        float t = 0.0f;
        #pragma unroll 8
        for (int c = 0; c < CD; ++c) t += df[c];
        atomicAdd(lossAcc, t);
    }
}

// --------------------------------------------------------------------------
// K4: commit_loss = COMMIT_W * (1 + BETA) * sum / (rows * cdim)
// --------------------------------------------------------------------------
__global__ void vq_finalize(const float* __restrict__ lossAcc,
                            float* __restrict__ lossOut) {
    if (threadIdx.x == 0 && blockIdx.x == 0)
        *lossOut = 1.25f * (*lossAcc) / ((float)NROWS * (float)CD);
}

// --------------------------------------------------------------------------
extern "C" void kernel_launch(void* const* d_in, const int* in_sizes, int n_in,
                              void* d_out, int out_size, void* d_ws, size_t ws_size,
                              hipStream_t stream) {
    const float* z   = (const float*)d_in[0];   // [8,1024,256]
    const float* Wq  = (const float*)d_in[1];   // [64,256]
    const float* bq  = (const float*)d_in[2];   // [64]
    const float* emb = (const float*)d_in[3];   // [16384,64]
    const float* Wp  = (const float*)d_in[4];   // [256,64]
    const float* bp  = (const float*)d_in[5];   // [256]

    char* ws = (char*)d_ws;
    float*          zpF   = (float*)         (ws);                              // 2 MB
    unsigned short* zpB   = (unsigned short*)(ws + ((size_t)2 << 20));          // 1 MB
    unsigned short* embB  = (unsigned short*)(ws + ((size_t)3 << 20));          // 2 MB
    float*          enorm = (float*)         (ws + ((size_t)5 << 20));          // 64 KB
    int*            idx   = (int*)           (ws + ((size_t)5 << 20) + (64u << 10)); // 32 KB
    float*          lossA = (float*)         (ws + ((size_t)5 << 20) + (96u << 10));

    float* out     = (float*)d_out;                 // [8,1024,256]
    float* idxOutF = out + (size_t)NROWS * DIMSZ;   // [8,1024] as float
    float* lossOut = idxOutF + NROWS;               // scalar

    vq_prep_emb<<<KCODES / 256, 256, 0, stream>>>(emb, embB, enorm, lossA);
    vq_proj_q  <<<NROWS / 4,    256, 0, stream>>>(z, Wq, bq, zpF, zpB);
    vq_argmin  <<<NROWS / 32,   256, 0, stream>>>(zpB, embB, enorm, idx, idxOutF);
    vq_out     <<<NROWS,        256, 0, stream>>>(emb, Wp, bp, zpF, idx, out, lossA);
    vq_finalize<<<1, 1, 0, stream>>>(lossA, lossOut);
}
